// MODEL_41910290874758
// MI455X (gfx1250) — matrix operand order repaired and checked by reading, not compiled, not run on hardware
//
#include <hip/hip_runtime.h>

typedef _Float16 f16;
typedef __attribute__((ext_vector_type(16))) _Float16 v16h;
typedef __attribute__((ext_vector_type(8)))  _Float16 v8h;
typedef __attribute__((ext_vector_type(8)))  float    v8f;
typedef __attribute__((ext_vector_type(4)))  float    v4f;

constexpr int DIMC = 128;   // embedding dim
constexpr int TC   = 64;    // triplets per entity
constexpr int BC   = 4096;  // batch
constexpr int NW   = 8;     // waves per block (wave32)

union AFrag { v16h v; v8h h8[2]; f16 h[16]; };

__device__ __forceinline__ float sigmoidf_(float x) { return 1.0f / (1.0f + __expf(-x)); }

// load 8 contiguous f32 and convert to 8 f16
__device__ __forceinline__ void cvt8(const float* __restrict__ p, f16* d) {
  v4f a = *(const v4f*)p;
  v4f b = *(const v4f*)(p + 4);
  d[0] = (f16)a[0]; d[1] = (f16)a[1]; d[2] = (f16)a[2]; d[3] = (f16)a[3];
  d[4] = (f16)b[0]; d[5] = (f16)b[1]; d[6] = (f16)b[2]; d[7] = (f16)b[3];
}

// ---------------------------------------------------------------------------
// Prep: convert MLP weights to f16, transposed to [N][K] so WMMA B-fragments
// are contiguous-K 32B loads per lane.
// ---------------------------------------------------------------------------
__global__ void prep_weights(const float* __restrict__ w1, const float* __restrict__ w2,
                             f16* __restrict__ w1T, f16* __restrict__ w2T) {
  int tid = blockIdx.x * blockDim.x + threadIdx.x;
  if (tid < 384 * 128) {
    int k = tid >> 7, n = tid & 127;
    w1T[n * 384 + k] = (f16)w1[tid];
  } else if (tid < 384 * 128 + 128 * 128) {
    int t2 = tid - 384 * 128;
    int k = t2 >> 7, n = t2 & 127;
    w2T[n * 128 + k] = (f16)w2[t2];
  }
}

// ---------------------------------------------------------------------------
// Attention layer: one wave per batch element, one (side, layer) per grid.yz.
// GEMM1 (16x384 @ 384x128): kk rolled, 8 live f32 accumulators, one transient
// A-fragment. An opaque *scalar offset* (volatile asm, SGPR) defeats LICM on
// the weight loads while keeping the pointers in the global address space so
// they lower to global_load_b128 (not flat_load, which also ties up DScnt).
// GEMM2 (16x128 @ 128x128): nt rolled, one accumulator, fused w3 GEMV.
// ---------------------------------------------------------------------------
__global__ __launch_bounds__(256) void attn_layer_kernel(
    const float* __restrict__ emb, const float* __restrict__ rel,
    const f16* __restrict__ w1T, const f16* __restrict__ w2T,
    const float* __restrict__ b1, const float* __restrict__ b2,
    const float* __restrict__ w3, const float* __restrict__ b3,
    const int* __restrict__ users, const int* __restrict__ items,
    const int* __restrict__ uh, const int* __restrict__ ur, const int* __restrict__ ut,
    const int* __restrict__ ih, const int* __restrict__ ir, const int* __restrict__ it,
    float* __restrict__ ei_out) {
  const int wave = threadIdx.x >> 5;
  const int lane = threadIdx.x & 31;
  const int hs   = lane >> 4;      // wave half select (K-chunk select for A/B frags)
  const int mrow = lane & 15;      // row within M-tile / column within N-tile
  const int hs8  = hs * 8;
  const int layer = blockIdx.y;
  const int side  = blockIdx.z;
  const int b = blockIdx.x * NW + wave;

  const int* idx0_arr = side ? items : users;
  const int* harr = side ? ih : uh;
  const int* rarr = side ? ir : ur;
  const int* tarr = side ? it : ut;

  const long base_bt = ((long)layer * BC + b) * TC;
  const float* e0row = emb + (long)idx0_arr[b] * DIMC;

  // per-lane weight base pointers; nt/kk displacements become immediate offsets
  const f16* w1base = w1T + mrow * 384 + hs * 16;
  const f16* w2base = w2T + mrow * DIMC + hs * 16;

  __shared__ f16   h1s[NW][16 * DIMC];  // per-wave hidden tile (f16)
  __shared__ float scs[NW][TC];         // per-wave attention scores / weights
  f16*   h1 = h1s[wave];
  float* sc = scs[wave];

  const float b3v = b3[0];

#pragma unroll 1
  for (int mt = 0; mt < 4; ++mt) {
    const int t = mt * 16 + mrow;
    const float* hrow = emb + (long)harr[base_bt + t] * DIMC;
    const float* rrow = rel + (long)rarr[base_bt + t] * DIMC;

    // Opaque scalar offset (always 0): redefined by volatile asm every mt
    // iteration, so weight-fragment loads are not loop-invariant and cannot
    // be hoisted/spilled, yet the pointers stay global-address-space.
    long woff = 0;
    asm volatile("" : "+s"(woff));
    const f16* w1p = w1base + woff;
    const f16* w2p = w2base + woff;

    __syncthreads();  // previous m-tile's h1 readers are done

    // ---- GEMM1: h1[16x128] = relu(concat(e0,h,r)[16x384] * W1 + b1) ----
    v8f acc[8];
#pragma unroll
    for (int nt = 0; nt < 8; ++nt) acc[nt] = (v8f){};

#pragma unroll 1
    for (int kk = 0; kk < 12; ++kk) {
      // build the A fragment for this 32-wide k-step (f32 gather -> f16)
      const float* basep = (kk < 4) ? e0row : (kk < 8) ? hrow : rrow;
      const int k0 = (kk & 3) * 32;
      AFrag a;
      cvt8(basep + k0 + hs * 16,     &a.h[0]);
      cvt8(basep + k0 + hs * 16 + 8, &a.h[8]);
#pragma unroll
      for (int nt = 0; nt < 8; ++nt) {
        v16h bf = *(const v16h*)(w1p + nt * (16 * 384) + kk * 32);
        acc[nt] = __builtin_amdgcn_wmma_f32_16x16x32_f16(false, a.v, false, bf,
                                                         (short)0, acc[nt], false, false);
      }
    }

#pragma unroll
    for (int nt = 0; nt < 8; ++nt) {
      const int n = nt * 16 + mrow;
      const float bn = b1[n];
#pragma unroll
      for (int r = 0; r < 8; ++r) {
        float v = acc[nt][r] + bn;
        v = v > 0.f ? v : 0.f;
        h1[(hs8 + r) * DIMC + n] = (f16)v;  // C layout: lane=col n, vgpr r=row
      }
    }

    __syncthreads();  // h1 visible

    // ---- GEMM2 A fragments straight from LDS (row-major h1) ----
    AFrag a2[4];
#pragma unroll
    for (int kk = 0; kk < 4; ++kk) {
      const int k0 = kk * 32;
      a2[kk].h8[0] = *(const v8h*)(h1 + mrow * DIMC + k0 + hs * 16);
      a2[kk].h8[1] = *(const v8h*)(h1 + mrow * DIMC + k0 + hs * 16 + 8);
    }

    float part[8];
#pragma unroll
    for (int r = 0; r < 8; ++r) part[r] = 0.f;

    // ---- GEMM2 + fused w3 GEMV: score[M] = sum_n relu(h2[M][n]) * w3[n] ----
#pragma unroll 1
    for (int nt = 0; nt < 8; ++nt) {
      const int n = nt * 16 + mrow;
      v8f acc2 = (v8f){};
#pragma unroll
      for (int kk = 0; kk < 4; ++kk) {
        v16h bf = *(const v16h*)(w2p + nt * (16 * DIMC) + kk * 32);
        acc2 = __builtin_amdgcn_wmma_f32_16x16x32_f16(false, a2[kk].v, false, bf,
                                                      (short)0, acc2, false, false);
      }
      const float bn = b2[n];
      const float wn = w3[n];
#pragma unroll
      for (int r = 0; r < 8; ++r) {
        float v = acc2[r] + bn;
        v = v > 0.f ? v : 0.f;
        part[r] += v * wn;
      }
    }

    // reduce over the 16 lanes sharing the same rows (xor masks stay in-half)
#pragma unroll
    for (int mask = 1; mask < 16; mask <<= 1) {
#pragma unroll
      for (int r = 0; r < 8; ++r) part[r] += __shfl_xor(part[r], mask, 32);
    }

    if (mrow == 0) {
#pragma unroll
      for (int r = 0; r < 8; ++r)
        sc[mt * 16 + hs8 + r] = sigmoidf_(part[r] + b3v);
    }
  }

  __syncthreads();

  // softmax over T=64 within the wave (2 scores per lane)
  float s0 = sc[lane], s1 = sc[lane + 32];
  float mx = fmaxf(s0, s1);
#pragma unroll
  for (int mask = 16; mask >= 1; mask >>= 1) mx = fmaxf(mx, __shfl_xor(mx, mask, 32));
  float ex0 = __expf(s0 - mx), ex1 = __expf(s1 - mx);
  float sum = ex0 + ex1;
#pragma unroll
  for (int mask = 16; mask >= 1; mask >>= 1) sum += __shfl_xor(sum, mask, 32);
  const float inv = 1.0f / sum;
  sc[lane]      = ex0 * inv;
  sc[lane + 32] = ex1 * inv;

  __syncthreads();

  // weighted sum of t-embeddings: ei[b][d] = sum_t w[t] * emb[t_idx][d]
  v4f acc4 = {};
  const int d0 = lane * 4;
#pragma unroll 1
  for (int tt = 0; tt < TC; ++tt) {
    const float wt = sc[tt];
    const float* trow = emb + (long)tarr[base_bt + tt] * DIMC;
    v4f v = *(const v4f*)(trow + d0);
    acc4 += wt * v;
  }
  const long slot = (long)(side * 2 + layer) * BC + b;
  *(v4f*)(ei_out + slot * DIMC + d0) = acc4;
}

// ---------------------------------------------------------------------------
// Aggregation: sequential decay-gated accumulation per side, then sigmoid(dot).
// One wave per batch element; decay MLP (256->128->1) done with VALU, x vector
// broadcast through per-wave LDS.
// ---------------------------------------------------------------------------
__global__ __launch_bounds__(256) void aggregate_kernel(
    const float* __restrict__ emb,
    const float* __restrict__ dw1, const float* __restrict__ dw2,
    const int* __restrict__ users, const int* __restrict__ items,
    const float* __restrict__ ei, float* __restrict__ out) {
  const int wave = threadIdx.x >> 5;
  const int lane = threadIdx.x & 31;
  const int b = blockIdx.x * NW + wave;
  __shared__ float xs_all[NW][2 * DIMC];
  float* xs = xs_all[wave];
  const int d0 = lane * 4;

  v4f eu = {}, ev = {};
#pragma unroll 1
  for (int side = 0; side < 2; ++side) {
    const int idx0 = side ? items[b] : users[b];
    v4f e = *(const v4f*)(emb + (long)idx0 * DIMC + d0);
#pragma unroll 1
    for (int layer = 0; layer < 2; ++layer) {
      const long slot = (long)(side * 2 + layer) * BC + b;
      v4f ei4 = *(const v4f*)(ei + slot * DIMC + d0);
      __syncthreads();
      *(v4f*)(xs + d0)        = e;     // accumulated e (aliasing semantics of ref)
      *(v4f*)(xs + DIMC + d0) = ei4;
      __syncthreads();
      v4f h = {};
#pragma unroll 1
      for (int k = 0; k < 2 * DIMC; ++k) {
        const float xk = xs[k];
        v4f w = *(const v4f*)(dw1 + k * DIMC + d0);
        h += xk * w;
      }
      v4f w2v = *(const v4f*)(dw2 + d0);
      float p = 0.f;
#pragma unroll
      for (int c = 0; c < 4; ++c) {
        float hv = h[c] > 0.f ? h[c] : 0.f;
        p += hv * w2v[c];
      }
#pragma unroll
      for (int mask = 16; mask >= 1; mask >>= 1) p += __shfl_xor(p, mask, 32);
      const float decay = sigmoidf_(p);
      e += decay * ei4;
    }
    if (side == 0) eu = e; else ev = e;
  }
  float d = 0.f;
#pragma unroll
  for (int c = 0; c < 4; ++c) d += eu[c] * ev[c];
#pragma unroll
  for (int mask = 16; mask >= 1; mask >>= 1) d += __shfl_xor(d, mask, 32);
  if (lane == 0) out[b] = sigmoidf_(d);
}

// ---------------------------------------------------------------------------
extern "C" void kernel_launch(void* const* d_in, const int* in_sizes, int n_in,
                              void* d_out, int out_size, void* d_ws, size_t ws_size,
                              hipStream_t stream) {
  const float* emb  = (const float*)d_in[0];
  const float* rel  = (const float*)d_in[1];
  const float* w1   = (const float*)d_in[2];
  const float* b1   = (const float*)d_in[3];
  const float* w2   = (const float*)d_in[4];
  const float* b2   = (const float*)d_in[5];
  const float* w3   = (const float*)d_in[6];
  const float* b3   = (const float*)d_in[7];
  const float* dw1  = (const float*)d_in[8];
  const float* dw2  = (const float*)d_in[9];
  const int* users  = (const int*)d_in[10];
  const int* items  = (const int*)d_in[11];
  const int* uh     = (const int*)d_in[12];
  const int* ur     = (const int*)d_in[13];
  const int* ut     = (const int*)d_in[14];
  const int* ih     = (const int*)d_in[15];
  const int* ir     = (const int*)d_in[16];
  const int* it     = (const int*)d_in[17];
  float* out = (float*)d_out;

  // workspace layout (256B-aligned base from hipMalloc):
  //   [0,            98304)  w1T  f16 [128][384]
  //   [98304,       131072)  w2T  f16 [128][128]
  //   [131072,  131072+8MB)  ei   f32 [4][4096][128]
  f16*   w1T = (f16*)d_ws;
  f16*   w2T = (f16*)((char*)d_ws + 98304);
  float* ei  = (float*)((char*)d_ws + 131072);

  prep_weights<<<(384 * 128 + 128 * 128 + 255) / 256, 256, 0, stream>>>(w1, w2, w1T, w2T);

  attn_layer_kernel<<<dim3(BC / NW, 2, 2), 256, 0, stream>>>(
      emb, rel, w1T, w2T, b1, b2, w3, b3,
      users, items, uh, ur, ut, ih, ir, it, ei);

  aggregate_kernel<<<BC / NW, 256, 0, stream>>>(emb, dw1, dw2, users, items, ei, out);
}
